// UpDownVec_80367428043299
// MI455X (gfx1250) — matrix operand order repaired and checked by reading, not compiled, not run on hardware
//
#include <hip/hip_runtime.h>

#define DIM 128
#define NTOT 299593
#define LEAFOFF 37449   // OFFS[6] = number of non-leaf (internal) nodes
#define NLEAF 262144

typedef unsigned int u32;
typedef unsigned short u16;
typedef __attribute__((ext_vector_type(16))) __bf16 v16bf;
typedef __attribute__((ext_vector_type(8)))  float  v8f;
typedef __attribute__((ext_vector_type(4)))  u32    u32x4;

union Frag { v16bf v; u32x4 q[2]; };

__device__ __forceinline__ u16 f2bf(float f) {        // f32 -> bf16 RNE
  u32 x = __float_as_uint(f);
  x += 0x7FFFu + ((x >> 16) & 1u);
  return (u16)(x >> 16);
}
__device__ __forceinline__ float relu_(float x) { return x > 0.0f ? x : 0.0f; }

// A fragment (16x32 bf16, MxK). ISA layout: lanes 0-15 row M=lane hold
// K = {0..7, 16..23}; lanes 16-31 hold K = {8..15, 24..31}.
__device__ __forceinline__ v16bf load_a_frag(const u16* __restrict__ base,
                                             int strideH, int lane, int kk) {
  const int r  = lane & 15;
  const int hi = (lane >> 4) & 1;
  const u16* p = base + r * strideH + kk * 32 + hi * 8;
  Frag f;
  f.q[0] = *(const u32x4*)(p);        // K = kk*32 + hi*8  + [0..7]
  f.q[1] = *(const u32x4*)(p + 16);   // K = kk*32 + 16 + hi*8 + [0..7]
  return f.v;
}

// B fragment (32x16 bf16, KxN), B[k][n] = W[n][k] with W row-major [N][Ktot].
// Lanes 0-15: col N=lane, K={0..7,16..23}; lanes 16-31: col N=lane-16, K={8..15,24..31}.
__device__ __forceinline__ v16bf load_b_frag(const u16* __restrict__ W,
                                             int strideH, int lane, int nbase, int kk) {
  const int col = (lane & 15) + nbase;
  const int hi  = (lane >> 4) & 1;
  const u16* p = W + col * strideH + kk * 32 + hi * 8;
  Frag f;
  f.q[0] = *(const u32x4*)(p);
  f.q[1] = *(const u32x4*)(p + 16);
  return f.v;
}

__device__ __forceinline__ v8f wmma_bf16(v16bf a, v16bf b, v8f c) {
  return __builtin_amdgcn_wmma_f32_16x16x32_bf16(false, a, false, b, (short)0, c,
                                                 false, false);
}

__device__ __forceinline__ float xor16_reduce(float s) {
  s += __int_as_float(__builtin_amdgcn_ds_swizzle(__float_as_int(s), (1 << 10) | 31));
  s += __int_as_float(__builtin_amdgcn_ds_swizzle(__float_as_int(s), (2 << 10) | 31));
  s += __int_as_float(__builtin_amdgcn_ds_swizzle(__float_as_int(s), (4 << 10) | 31));
  s += __int_as_float(__builtin_amdgcn_ds_swizzle(__float_as_int(s), (8 << 10) | 31));
  return s;
}

// ---------------------------------------------------------------------------
// Convert the 4 weight matrices to bf16 (row-major) and set root ctx = 1.0.
__global__ __launch_bounds__(256) void prep_kernel(
    const float* __restrict__ Ws, const float* __restrict__ Wc,
    const float* __restrict__ Wx, const float* __restrict__ Wf,
    u16* __restrict__ WsB, u16* __restrict__ WcB,
    u16* __restrict__ WxB, u16* __restrict__ WfB,
    float* __restrict__ ctxRoot) {
  int i = blockIdx.x * 256 + threadIdx.x;
  if (i < 16384)      WsB[i]         = f2bf(Ws[i]);
  else if (i < 32768) WcB[i - 16384] = f2bf(Wc[i - 16384]);
  else if (i < 49152) WxB[i - 32768] = f2bf(Wx[i - 32768]);
  else if (i < 81920) WfB[i - 49152] = f2bf(Wf[i - 49152]);
  if (i < DIM) ctxRoot[i] = 1.0f;
}

// ---------------------------------------------------------------------------
// Up step for one level: agg[l] = init + mean_T relu(Wc @ mean_S relu(Ws @ ch))
// Block = 16 parents = 128 child rows.  8 waves, each owns a 16-row strip.
__global__ __launch_bounds__(256) void up_level_kernel(
    const float* __restrict__ childAgg, const float* __restrict__ initV,
    float* __restrict__ aggDst,
    const u16* __restrict__ WsB, const float* __restrict__ bs,
    const u16* __restrict__ WcB, const float* __restrict__ bc,
    int nParent) {
  __shared__ u16 Xs[128 * 136];   // child rows bf16 (stride 136: conflict-free)
  __shared__ u16 Sib[32 * 136];   // sibling means bf16
  const int tid  = threadIdx.x;
  const int lane = tid & 31;
  const int wave = tid >> 5;
  const int r15  = lane & 15;
  const int hi   = (lane >> 4) & 1;
  const int pBase  = blockIdx.x * 16;
  const int nChild = nParent * 8;
  const int cBase  = pBase * 8;

  { // stage 128 child rows -> bf16 LDS (clamped for ragged tail)
    int r  = tid >> 1;
    int c0 = (tid & 1) * 64;
    int grow = cBase + r;
    if (grow >= nChild) grow = nChild - 1;
    const float* src = childAgg + (long)grow * DIM + c0;
    u16* dst = &Xs[r * 136 + c0];
#pragma unroll
    for (int i = 0; i < 64; i += 2)
      *(u32*)(dst + i) = (u32)f2bf(src[i]) | ((u32)f2bf(src[i + 1]) << 16);
  }
  __syncthreads();

  { // GEMM1: relu(child @ Ws^T + bs) then mean over sibling groups of 4 rows.
    const int strip = wave;
    const u16* aBase = &Xs[strip * 16 * 136];
#pragma unroll 1
    for (int nt = 0; nt < 8; ++nt) {
      v8f acc = {};
#pragma unroll
      for (int kk = 0; kk < 4; ++kk) {
        v16bf a = load_a_frag(aBase, 136, lane, kk);
        v16bf b = load_b_frag(WsB, DIM, lane, nt * 16, kk);
        acc = wmma_bf16(a, b, acc);
      }
      const int n = nt * 16 + r15;
      const float bias = bs[n];
      float g0 = 0.f, g1 = 0.f;
#pragma unroll
      for (int r = 0; r < 4; ++r) g0 += relu_(acc[r] + bias);
#pragma unroll
      for (int r = 4; r < 8; ++r) g1 += relu_(acc[r] + bias);
      // VGPR r <-> tile row r + 8*hi; sibling groups = 4 consecutive rows.
      const int sr = strip * 4 + hi * 2;
      Sib[(sr + 0) * 136 + n] = f2bf(g0 * 0.25f);
      Sib[(sr + 1) * 136 + n] = f2bf(g1 * 0.25f);
    }
  }
  __syncthreads();

  { // GEMM2: relu(sib @ Wc^T + bc), mean over tactic pairs, + init -> aggDst
#pragma unroll 1
    for (int t = wave; t < 16; t += 8) {
      const int strip2 = t >> 3;
      const int nt     = t & 7;
      const u16* aBase = &Sib[strip2 * 16 * 136];
      v8f acc = {};
#pragma unroll
      for (int kk = 0; kk < 4; ++kk) {
        v16bf a = load_a_frag(aBase, 136, lane, kk);
        v16bf b = load_b_frag(WcB, DIM, lane, nt * 16, kk);
        acc = wmma_bf16(a, b, acc);
      }
      const int n = nt * 16 + r15;
      const float bias = bc[n];
#pragma unroll
      for (int r = 0; r < 8; r += 2) {
        float v0 = relu_(acc[r] + bias);
        float v1 = relu_(acc[r + 1] + bias);
        float m  = 0.5f * (v0 + v1);          // EPS == 1, mean over T=2
        int pLoc = (strip2 * 16 + hi * 8 + r) >> 1;
        int p = pBase + pLoc;
        if (p < nParent) {
          long idx = (long)p * DIM + n;
          aggDst[idx] = initV[idx] + m;
        }
      }
    }
  }
}

// ---------------------------------------------------------------------------
// Down step for internal levels (children are internal nodes): writes ctx rows.
// ctx_child = (relu(Wx@ctx_par) + sum_group relu(Wx@agg_ch) - relu(Wx@agg_self)) / 4
__global__ __launch_bounds__(256) void down_level_kernel(
    const float* __restrict__ parentCtx, const float* __restrict__ childAgg,
    float* __restrict__ ctxDst,
    const u16* __restrict__ WxB, const float* __restrict__ bx,
    int nParent) {
  __shared__ u16  Xs[128 * 136];   // child agg bf16
  __shared__ u16  Ps[16 * 136];    // parent ctx bf16
  __shared__ float CtxH[16 * 132]; // relu(parentCtx @ Wx^T + bx)
  const int tid = threadIdx.x, lane = tid & 31, wave = tid >> 5;
  const int r15 = lane & 15, hi = (lane >> 4) & 1;
  const int pBase  = blockIdx.x * 16;
  const int nChild = nParent * 8;
  const int cBase  = pBase * 8;

  { // children -> LDS bf16
    int r = tid >> 1, c0 = (tid & 1) * 64;
    int grow = cBase + r;
    if (grow >= nChild) grow = nChild - 1;
    const float* src = childAgg + (long)grow * DIM + c0;
    u16* dst = &Xs[r * 136 + c0];
#pragma unroll
    for (int i = 0; i < 64; i += 2)
      *(u32*)(dst + i) = (u32)f2bf(src[i]) | ((u32)f2bf(src[i + 1]) << 16);
  }
  { // parents -> LDS bf16 (16 rows, 16 threads per row)
    int r = tid >> 4, c0 = (tid & 15) * 8;
    int grow = pBase + r;
    if (grow >= nParent) grow = nParent - 1;
    const float* src = parentCtx + (long)grow * DIM + c0;
    u16* dst = &Ps[r * 136 + c0];
#pragma unroll
    for (int i = 0; i < 8; i += 2)
      *(u32*)(dst + i) = (u32)f2bf(src[i]) | ((u32)f2bf(src[i + 1]) << 16);
  }
  __syncthreads();

  { // CtxH GEMM: one 16-row strip, wave -> N-tile
    const int nt = wave;
    v8f acc = {};
#pragma unroll
    for (int kk = 0; kk < 4; ++kk) {
      v16bf a = load_a_frag(Ps, 136, lane, kk);
      v16bf b = load_b_frag(WxB, DIM, lane, nt * 16, kk);
      acc = wmma_bf16(a, b, acc);
    }
    const int n = nt * 16 + r15;
    const float bias = bx[n];
#pragma unroll
    for (int r = 0; r < 8; ++r)
      CtxH[(r + 8 * hi) * 132 + n] = relu_(acc[r] + bias);
  }
  __syncthreads();

  { // child h GEMM + combine, all in registers
    const int strip = wave;
    const u16* aBase = &Xs[strip * 16 * 136];
#pragma unroll 1
    for (int nt = 0; nt < 8; ++nt) {
      v8f acc = {};
#pragma unroll
      for (int kk = 0; kk < 4; ++kk) {
        v16bf a = load_a_frag(aBase, 136, lane, kk);
        v16bf b = load_b_frag(WxB, DIM, lane, nt * 16, kk);
        acc = wmma_bf16(a, b, acc);
      }
      const int n = nt * 16 + r15;
      const float bias = bx[n];
      float h[8];
#pragma unroll
      for (int r = 0; r < 8; ++r) h[r] = relu_(acc[r] + bias);
      const float gs0 = h[0] + h[1] + h[2] + h[3];   // sibling-group sums
      const float gs1 = h[4] + h[5] + h[6] + h[7];
      const float ch  = CtxH[(strip * 2 + hi) * 132 + n];  // parent of this half
#pragma unroll
      for (int r = 0; r < 8; ++r) {
        float o = (ch + ((r < 4) ? gs0 : gs1) - h[r]) * 0.25f;
        int crow = cBase + strip * 16 + hi * 8 + r;
        if (crow < nChild) ctxDst[(long)crow * DIM + n] = o;
      }
    }
  }
}

// ---------------------------------------------------------------------------
// Leaf level, fully fused: down-step for leaves + final head, so the 134 MB
// leaf-ctx tensor never touches HBM.  Leaf agg == leaf initial vecs (in Xs);
// leaf ctx lands in Cs (bf16) and both feed the K=256 Wf GEMM directly.
__global__ __launch_bounds__(256) void down_leaf_final_kernel(
    const float* __restrict__ parentCtx, const float* __restrict__ leafInit,
    const u16* __restrict__ WxB, const float* __restrict__ bx,
    const u16* __restrict__ WfB, const float* __restrict__ bfv,
    const float* __restrict__ Wh, const float* __restrict__ bh,
    float* __restrict__ outLeaf, int nParent) {
  __shared__ u16  Xs[128 * 136];   // leaf agg bf16  (= final A, K=128..255 half)
  __shared__ u16  Cs[128 * 136];   // leaf ctx bf16  (= final A, K=0..127 half)
  __shared__ u16  Ps[16 * 136];    // parent ctx bf16
  __shared__ float CtxH[16 * 132];
  const int tid = threadIdx.x, lane = tid & 31, wave = tid >> 5;
  const int r15 = lane & 15, hi = (lane >> 4) & 1;
  const int pBase  = blockIdx.x * 16;
  const int nChild = nParent * 8;
  const int cBase  = pBase * 8;

  { // leaves -> LDS bf16
    int r = tid >> 1, c0 = (tid & 1) * 64;
    int grow = cBase + r;
    if (grow >= nChild) grow = nChild - 1;
    const float* src = leafInit + (long)grow * DIM + c0;
    u16* dst = &Xs[r * 136 + c0];
#pragma unroll
    for (int i = 0; i < 64; i += 2)
      *(u32*)(dst + i) = (u32)f2bf(src[i]) | ((u32)f2bf(src[i + 1]) << 16);
  }
  { // parents -> LDS bf16
    int r = tid >> 4, c0 = (tid & 15) * 8;
    int grow = pBase + r;
    if (grow >= nParent) grow = nParent - 1;
    const float* src = parentCtx + (long)grow * DIM + c0;
    u16* dst = &Ps[r * 136 + c0];
#pragma unroll
    for (int i = 0; i < 8; i += 2)
      *(u32*)(dst + i) = (u32)f2bf(src[i]) | ((u32)f2bf(src[i + 1]) << 16);
  }
  __syncthreads();

  { // CtxH = relu(parentCtx @ Wx^T + bx)
    const int nt = wave;
    v8f acc = {};
#pragma unroll
    for (int kk = 0; kk < 4; ++kk) {
      v16bf a = load_a_frag(Ps, 136, lane, kk);
      v16bf b = load_b_frag(WxB, DIM, lane, nt * 16, kk);
      acc = wmma_bf16(a, b, acc);
    }
    const int n = nt * 16 + r15;
    const float bias = bx[n];
#pragma unroll
    for (int r = 0; r < 8; ++r)
      CtxH[(r + 8 * hi) * 132 + n] = relu_(acc[r] + bias);
  }
  __syncthreads();

  { // h GEMM + combine; ctx goes straight to LDS as bf16 (no HBM store)
    const int strip = wave;
    const u16* aBase = &Xs[strip * 16 * 136];
#pragma unroll 1
    for (int nt = 0; nt < 8; ++nt) {
      v8f acc = {};
#pragma unroll
      for (int kk = 0; kk < 4; ++kk) {
        v16bf a = load_a_frag(aBase, 136, lane, kk);
        v16bf b = load_b_frag(WxB, DIM, lane, nt * 16, kk);
        acc = wmma_bf16(a, b, acc);
      }
      const int n = nt * 16 + r15;
      const float bias = bx[n];
      float h[8];
#pragma unroll
      for (int r = 0; r < 8; ++r) h[r] = relu_(acc[r] + bias);
      const float gs0 = h[0] + h[1] + h[2] + h[3];
      const float gs1 = h[4] + h[5] + h[6] + h[7];
      const float ch  = CtxH[(strip * 2 + hi) * 132 + n];
#pragma unroll
      for (int r = 0; r < 8; ++r) {
        float o = (ch + ((r < 4) ? gs0 : gs1) - h[r]) * 0.25f;
        Cs[(strip * 16 + hi * 8 + r) * 136 + n] = f2bf(o);
      }
    }
  }
  __syncthreads();

  { // final head: relu([ctx|agg] @ Wf^T + bf) . Wh + bh, K=256 as two K=128.
    const int strip = wave;
    const u16* aCtx = &Cs[strip * 16 * 136];
    const u16* aAgg = &Xs[strip * 16 * 136];
    float wsum[8] = {0, 0, 0, 0, 0, 0, 0, 0};
#pragma unroll 1
    for (int nt = 0; nt < 8; ++nt) {
      v8f acc = {};
#pragma unroll
      for (int kk = 0; kk < 4; ++kk) {      // Wf columns 0..127 (ctx half)
        v16bf a = load_a_frag(aCtx, 136, lane, kk);
        v16bf b = load_b_frag(WfB, 2 * DIM, lane, nt * 16, kk);
        acc = wmma_bf16(a, b, acc);
      }
#pragma unroll
      for (int kk = 0; kk < 4; ++kk) {      // Wf columns 128..255 (agg half)
        v16bf a = load_a_frag(aAgg, 136, lane, kk);
        v16bf b = load_b_frag(WfB + DIM, 2 * DIM, lane, nt * 16, kk);
        acc = wmma_bf16(a, b, acc);
      }
      const int n = nt * 16 + r15;
      const float bias = bfv[n];
      const float w = Wh[n];
#pragma unroll
      for (int r = 0; r < 8; ++r) wsum[r] += relu_(acc[r] + bias) * w;
    }
#pragma unroll
    for (int r = 0; r < 8; ++r) wsum[r] = xor16_reduce(wsum[r]);
    if (r15 == 0) {
      const float bias = bh[0];
#pragma unroll
      for (int r = 0; r < 8; ++r) {
        int crow = cBase + strip * 16 + hi * 8 + r;
        if (crow < nChild) outLeaf[crow] = wsum[r] + bias;
      }
    }
  }
}

// ---------------------------------------------------------------------------
// Final head for the 37,449 internal nodes (ctx + agg both in workspace).
__global__ __launch_bounds__(256) void final_internal_kernel(
    const float* __restrict__ ctxAll, const float* __restrict__ aggWs,
    const u16* __restrict__ WfB, const float* __restrict__ bfv,
    const float* __restrict__ Wh, const float* __restrict__ bh,
    float* __restrict__ out, int nRows) {
  __shared__ u16   Xs[64 * 264];   // [ctx(128) | agg(128)] bf16 per row
  __shared__ float rsum[64];
  const int tid = threadIdx.x, lane = tid & 31, wave = tid >> 5;
  const int r15 = lane & 15, hi = (lane >> 4) & 1;
  const long rBase = (long)blockIdx.x * 64;

  { // stage: 4 threads per row (seg 0,1 = ctx halves; seg 2,3 = agg halves)
    int r = tid >> 2, seg = tid & 3;
    long row = rBase + r;
    if (row >= nRows) row = nRows - 1;
    const float* src = (seg < 2) ? (ctxAll + row * DIM + seg * 64)
                                 : (aggWs + row * DIM + (seg - 2) * 64);
    u16* dst = &Xs[r * 264 + seg * 64];
#pragma unroll
    for (int i = 0; i < 64; i += 2)
      *(u32*)(dst + i) = (u32)f2bf(src[i]) | ((u32)f2bf(src[i + 1]) << 16);
  }
  if (tid < 64) rsum[tid] = 0.0f;
  __syncthreads();

  const int strip  = wave >> 1;           // 4 strips of 16 rows
  const int ntBase = (wave & 1) * 4;      // 2 waves split the 8 N-tiles
  const u16* aBase = &Xs[strip * 16 * 264];
  float wsum[8] = {0, 0, 0, 0, 0, 0, 0, 0};
#pragma unroll 1
  for (int nt = ntBase; nt < ntBase + 4; ++nt) {
    v8f acc = {};
#pragma unroll
    for (int kk = 0; kk < 8; ++kk) {      // K = 256
      v16bf a = load_a_frag(aBase, 264, lane, kk);
      v16bf b = load_b_frag(WfB, 2 * DIM, lane, nt * 16, kk);
      acc = wmma_bf16(a, b, acc);
    }
    const int n = nt * 16 + r15;
    const float bias = bfv[n];
    const float w = Wh[n];
#pragma unroll
    for (int r = 0; r < 8; ++r) wsum[r] += relu_(acc[r] + bias) * w;
  }
#pragma unroll
  for (int r = 0; r < 8; ++r) wsum[r] = xor16_reduce(wsum[r]);
  if (r15 == 0) {
#pragma unroll
    for (int r = 0; r < 8; ++r)
      atomicAdd(&rsum[strip * 16 + hi * 8 + r], wsum[r]);
  }
  __syncthreads();
  if (tid < 64) {
    long row = rBase + tid;
    if (row < nRows) out[row] = rsum[tid] + bh[0];
  }
}

// ---------------------------------------------------------------------------
extern "C" void kernel_launch(void* const* d_in, const int* in_sizes, int n_in,
                              void* d_out, int out_size, void* d_ws, size_t ws_size,
                              hipStream_t stream) {
  (void)in_sizes; (void)n_in; (void)out_size; (void)ws_size;
  static const long OFFS[8] = {0, 1, 9, 73, 585, 4681, 37449, 299593};
  static const int  SIZ[7]  = {1, 8, 64, 512, 4096, 32768, 262144};

  const float* initial = (const float*)d_in[0];
  const float* Ws  = (const float*)d_in[1];
  const float* bs  = (const float*)d_in[2];
  const float* Wc  = (const float*)d_in[3];
  const float* bc  = (const float*)d_in[4];
  const float* Wx  = (const float*)d_in[5];
  const float* bx  = (const float*)d_in[6];
  const float* Wf  = (const float*)d_in[7];
  const float* bfv = (const float*)d_in[8];
  const float* Wh  = (const float*)d_in[9];
  const float* bh  = (const float*)d_in[10];
  float* out = (float*)d_out;

  // Workspace: ctx + agg for internal nodes only (leaf ctx never hits HBM).
  // Total = 2 * 37449 * 512 B + 160 KB  ~= 38.5 MB.
  float* ctxAll = (float*)d_ws;                         // LEAFOFF*128 f32
  float* aggWs  = ctxAll + (long)LEAFOFF * DIM;         // LEAFOFF*128 f32
  u16*   WsB    = (u16*)(aggWs + (long)LEAFOFF * DIM);
  u16*   WcB    = WsB + 16384;
  u16*   WxB    = WcB + 16384;
  u16*   WfB    = WxB + 16384;

  prep_kernel<<<320, 256, 0, stream>>>(Ws, Wc, Wx, Wf, WsB, WcB, WxB, WfB, ctxAll);

  for (int l = 5; l >= 0; --l) {   // bottom-up
    const float* child = (l == 5) ? (initial + OFFS[6] * DIM)
                                  : (aggWs + OFFS[l + 1] * DIM);
    dim3 grid((SIZ[l] + 15) / 16);
    up_level_kernel<<<grid, 256, 0, stream>>>(child, initial + OFFS[l] * DIM,
                                              aggWs + OFFS[l] * DIM,
                                              WsB, bs, WcB, bc, SIZ[l]);
  }
  for (int l = 0; l <= 4; ++l) {   // top-down over internal children
    dim3 grid((SIZ[l] + 15) / 16);
    down_level_kernel<<<grid, 256, 0, stream>>>(ctxAll + OFFS[l] * DIM,
                                                aggWs + OFFS[l + 1] * DIM,
                                                ctxAll + OFFS[l + 1] * DIM,
                                                WxB, bx, SIZ[l]);
  }
  // Leaf level: down-step + final head fused (saves ~270 MB of HBM traffic).
  down_leaf_final_kernel<<<SIZ[5] / 16, 256, 0, stream>>>(
      ctxAll + OFFS[5] * DIM, initial + OFFS[6] * DIM,
      WxB, bx, WfB, bfv, Wh, bh, out + OFFS[6], SIZ[5]);
  // Final head for internal nodes.
  final_internal_kernel<<<(LEAFOFF + 63) / 64, 256, 0, stream>>>(
      ctxAll, aggWs, WfB, bfv, Wh, bh, out, LEAFOFF);
}
